// FPE_16_5291399709492
// MI455X (gfx1250) — matrix operand order repaired and checked by reading, not compiled, not run on hardware
//
#include <hip/hip_runtime.h>
#include <cstdint>

#define BATCH 65536
#define NW 3
#define NF 48
#define NH 16
#define NL 10
#define E_DIM 97
#define E3 291           // 3*E
#define KPAD 128         // padded concat dim (4 K-tiles of 32)
#define NPAD_QKV 304     // padded 3E      (19 N-tiles of 16)
#define XA_STRIDE 320    // padded 291     (10 K-tiles of 32)
#define LAT 160          // NH*NL

typedef __attribute__((ext_vector_type(16))) __bf16 v16bf;
typedef __attribute__((ext_vector_type(8)))  float  v8f;
typedef __attribute__((ext_vector_type(4)))  int    v4i;
typedef __attribute__((ext_vector_type(8)))  int    v8i;

union Frag { v16bf v; unsigned int u[8]; };

__device__ __forceinline__ unsigned short f2bf(float f) {
  union { float f; unsigned u; } x; x.f = f;
  unsigned u = x.u;
  unsigned r = u + 0x7fffu + ((u >> 16) & 1u);   // round-to-nearest-even
  return (unsigned short)(r >> 16);
}
__device__ __forceinline__ float sigmoidf_(float x) { return 1.0f / (1.0f + __expf(-x)); }

// ---------------------------------------------------------------------------
// Tensor Data Mover: 1-D contiguous global -> LDS tile load.
// D# per CDNA5 ISA (§8.3 group0, §8.4 group1):
//   group0: count=1 | lds_addr | global_addr[56:0] | type=2
//   group1: data_size=2B, tile_dim0 = tensor_dim0 = stride0 = nelem (1-D tile)
// Issued by one wave per workgroup; completion via TENSORcnt.
// ---------------------------------------------------------------------------
__device__ __forceinline__ void tdm_load_1d(const void* gsrc, unsigned lds_byte_off,
                                            unsigned nelem /* 2-byte elements */) {
  unsigned long long ga = (unsigned long long)(uintptr_t)gsrc;
  v4i g0;
  g0[0] = 1;                                            // count=1, user mode
  g0[1] = (int)lds_byte_off;                            // lds_addr
  g0[2] = (int)(ga & 0xffffffffu);                      // global_addr[31:0]
  g0[3] = (int)(((ga >> 32) & 0x01ffffffu) | 0x80000000u); // addr[56:32] | type=2
  v8i g1;
  g1[0] = 0x00010000;                                   // wg_mask=0, data_size=1 (2B)
  g1[1] = (int)((nelem & 0xffffu) << 16);               // tensor_dim0[15:0] @ bits 63:48
  g1[2] = (int)(((nelem >> 16) & 0xffffu) | (1u << 16)); // dim0[31:16], tensor_dim1=1
  g1[3] = (int)((nelem & 0xffffu) << 16);               // dim1[31:16]=0, tile_dim0 @127:112
  g1[4] = 0;                                            // tile_dim1=0, tile_dim2=0 (1-D)
  g1[5] = (int)nelem;                                   // tensor_dim0_stride[31:0]
  g1[6] = 0;                                            // stride0[47:32]=0, stride1 lo=0
  g1[7] = 0;
  asm volatile("tensor_load_to_lds %0, %1" :: "s"(g0), "s"(g1) : "memory");
}

// ---------------------------------------------------------------------------
// Kernel 0: weight prep. bf16-pack padded QKV weights; build fused
// (out-proj ∘ encoder) weight W_comb and fused bias.
// ---------------------------------------------------------------------------
__global__ void k_prep(const float* __restrict__ mha_in_w,
                       const float* __restrict__ mha_out_w,
                       const float* __restrict__ mha_out_b,
                       const float* __restrict__ enc_w,
                       const float* __restrict__ enc_b,
                       unsigned short* __restrict__ W1,
                       unsigned short* __restrict__ Wc,
                       float* __restrict__ bcomb) {
  int gid = blockIdx.x * blockDim.x + threadIdx.x;
  int nth = gridDim.x * blockDim.x;
  for (int idx = gid; idx < NPAD_QKV * KPAD; idx += nth) {
    int n = idx >> 7, k = idx & 127;
    W1[idx] = (n < E3 && k < E_DIM) ? f2bf(mha_in_w[n * E_DIM + k]) : (unsigned short)0;
  }
  for (int idx = gid; idx < LAT * XA_STRIDE; idx += nth) {
    int j = idx / XA_STRIDE, kk = idx - j * XA_STRIDE;
    unsigned short o = 0;
    if (kk < E3) {
      int l = kk / E_DIM, e = kk - l * E_DIM;
      float s = 0.f;
      for (int d = 0; d < E_DIM; ++d)
        s += enc_w[j * E3 + l * E_DIM + d] * mha_out_w[d * E_DIM + e];
      o = f2bf(s);
    }
    Wc[idx] = o;
  }
  for (int j = gid; j < LAT; j += nth) {
    float s = enc_b[j];
    for (int kk = 0; kk < E3; ++kk) s += enc_w[j * E3 + kk] * mha_out_b[kk % E_DIM];
    bcomb[j] = s;
  }
}

// ---------------------------------------------------------------------------
// Kernel 1: GRU scan + GAT hub per element; emit K-padded bf16 concat rows.
// Row (b*3+w): cols 0..47 = GRU hidden trace, col 96 = GAT hub[w], else 0.
// ---------------------------------------------------------------------------
__global__ void k_gru_gat(const float* __restrict__ t, const float* __restrict__ h0,
                          const float* __restrict__ Wih, const float* __restrict__ Whh,
                          const float* __restrict__ bih, const float* __restrict__ bhh,
                          const float* __restrict__ gat_fc, const float* __restrict__ gat_attn,
                          unsigned short* __restrict__ Xc) {
  int b = blockIdx.x * blockDim.x + threadIdx.x;
  if (b >= BATCH) return;
  const float* tb = t + (size_t)b * NW * NF;
  float h[NW];
#pragma unroll
  for (int w = 0; w < NW; ++w) h[w] = h0[b * NW + w];
  unsigned short* row0 = Xc + (size_t)(b * NW) * KPAD;

  for (int f = 0; f < NF; ++f) {
    float x0 = tb[0 * NF + f], x1 = tb[1 * NF + f], x2 = tb[2 * NF + f];
    float hn[NW];
#pragma unroll
    for (int i = 0; i < NW; ++i) {
      float gxr = bih[i]     + Wih[i * 3 + 0] * x0 + Wih[i * 3 + 1] * x1 + Wih[i * 3 + 2] * x2;
      float gxz = bih[3 + i] + Wih[(3 + i) * 3 + 0] * x0 + Wih[(3 + i) * 3 + 1] * x1 + Wih[(3 + i) * 3 + 2] * x2;
      float gxn = bih[6 + i] + Wih[(6 + i) * 3 + 0] * x0 + Wih[(6 + i) * 3 + 1] * x1 + Wih[(6 + i) * 3 + 2] * x2;
      float ghr = bhh[i]     + Whh[i * 3 + 0] * h[0] + Whh[i * 3 + 1] * h[1] + Whh[i * 3 + 2] * h[2];
      float ghz = bhh[3 + i] + Whh[(3 + i) * 3 + 0] * h[0] + Whh[(3 + i) * 3 + 1] * h[1] + Whh[(3 + i) * 3 + 2] * h[2];
      float ghn = bhh[6 + i] + Whh[(6 + i) * 3 + 0] * h[0] + Whh[(6 + i) * 3 + 1] * h[1] + Whh[(6 + i) * 3 + 2] * h[2];
      float r = sigmoidf_(gxr + ghr);
      float z = sigmoidf_(gxz + ghz);
      float n = tanhf(gxn + r * ghn);
      hn[i] = (1.0f - z) * n + z * h[i];
    }
    h[0] = hn[0]; h[1] = hn[1]; h[2] = hn[2];
    row0[0 * KPAD + f] = f2bf(h[0]);
    row0[1 * KPAD + f] = f2bf(h[1]);
    row0[2 * KPAD + f] = f2bf(h[2]);
  }
  // GAT: dst node is all-zero so its attention term vanishes.
  float a0 = gat_attn[0], a1 = gat_attn[1], a2 = gat_attn[2];
  float e[NF];
  float emax = -1e30f;
  for (int j = 0; j < NF; ++j) {
    float n0 = tb[0 * NF + j], n1 = tb[1 * NF + j], n2 = tb[2 * NF + j];
    float z0 = gat_fc[0] * n0 + gat_fc[1] * n1 + gat_fc[2] * n2;
    float z1 = gat_fc[3] * n0 + gat_fc[4] * n1 + gat_fc[5] * n2;
    float z2 = gat_fc[6] * n0 + gat_fc[7] * n1 + gat_fc[8] * n2;
    float s = z0 * a0 + z1 * a1 + z2 * a2;
    s = (s > 0.f) ? s : 0.2f * s;          // leaky_relu 0.2
    e[j] = s;
    emax = fmaxf(emax, s);
  }
  float esum = 0.f;
  for (int j = 0; j < NF; ++j) { e[j] = __expf(e[j] - emax); esum += e[j]; }
  float inv = 1.0f / esum;
  float hub0 = 0.f, hub1 = 0.f, hub2 = 0.f;
  for (int j = 0; j < NF; ++j) {
    float n0 = tb[0 * NF + j], n1 = tb[1 * NF + j], n2 = tb[2 * NF + j];
    float z0 = gat_fc[0] * n0 + gat_fc[1] * n1 + gat_fc[2] * n2;
    float z1 = gat_fc[3] * n0 + gat_fc[4] * n1 + gat_fc[5] * n2;
    float z2 = gat_fc[6] * n0 + gat_fc[7] * n1 + gat_fc[8] * n2;
    float al = e[j] * inv;
    hub0 += al * z0; hub1 += al * z1; hub2 += al * z2;
  }
  float hub[NW] = {hub0, hub1, hub2};
#pragma unroll
  for (int w = 0; w < NW; ++w) {
    unsigned short* r = row0 + w * KPAD;
    for (int f = NF; f < KPAD; ++f) r[f] = 0;
    r[2 * NF] = f2bf(hub[w]);              // col 96 = hub
  }
}

// ---------------------------------------------------------------------------
// Kernel 2: QKV GEMM (WMMA bf16) + fused 3x3 attention per 16-element tile.
// 128 threads (4 waves). LDS: X tile (48x128 bf16, TDM-staged) + QKV (48x292 f32).
// ---------------------------------------------------------------------------
__global__ void k_qkv_attn(const unsigned short* __restrict__ Xc,
                           const unsigned short* __restrict__ W1,
                           const float* __restrict__ bqkv,
                           unsigned short* __restrict__ Xa) {
  extern __shared__ char smem2[];
  unsigned short* lds_x = (unsigned short*)smem2;            // 48*128*2 = 12288 B @ LDS 0
  float* lds_qkv = (float*)(smem2 + 48 * KPAD * 2);          // 48*292*4 = 56064 B
  const int QS = 292;
  int tid = threadIdx.x;
  int elemBase = blockIdx.x * 16;

  // TDM-stage X tile into LDS (one descriptor per workgroup, wave 0 issues)
  if (tid < 32) {
    tdm_load_1d(Xc + (size_t)(elemBase * 3) * KPAD, 0u, 48u * KPAD);
    __builtin_amdgcn_s_wait_tensorcnt(0);
  }
  __syncthreads();

  int lane = tid & 31;
  int wave = tid >> 5;
  int hf = lane >> 4;      // lane half
  int nlo = lane & 15;

  for (int tile = wave; tile < 3 * 19; tile += 4) {
    int mt = tile / 19, nt = tile % 19;
    int nrow = nt * 16 + nlo;
    __builtin_prefetch(&W1[(size_t)nrow * KPAD], 0, 1);
    v8f acc = {0.f, 0.f, 0.f, 0.f, 0.f, 0.f, 0.f, 0.f};
#pragma unroll
    for (int kt = 0; kt < 4; ++kt) {
      int kb = kt * 32;
      Frag A, Bf;
#pragma unroll
      for (int v = 0; v < 8; ++v) {
        int ka = kb + 2 * (v & 3) + 8 * hf + 16 * (v >> 2);          // A packing
        A.u[v] = *(const unsigned int*)&lds_x[(mt * 16 + nlo) * KPAD + ka];
        int kbcol = kb + 2 * (v & 3) + 8 * hf + 16 * (v >> 2);       // B packing
        Bf.u[v] = *(const unsigned int*)&W1[(size_t)nrow * KPAD + kbcol];
      }
      acc = __builtin_amdgcn_wmma_f32_16x16x32_bf16(false, A.v, false, Bf.v,
                                                    (short)0, acc, false, false);
    }
    int col = nt * 16 + nlo;
    if (col < E3) {
      float bias = bqkv[col];
#pragma unroll
      for (int j = 0; j < 8; ++j) {
        int row = mt * 16 + j + 8 * hf;
        lds_qkv[row * QS + col] = acc[j] + bias;
      }
    }
  }
  __syncthreads();

  // 3x3 attention per (element, query-token)
  if (tid < 48) {
    int e = tid / 3, i = tid - e * 3;
    const float* q = &lds_qkv[(e * 3 + i) * QS];
    const float scale = 0.10153461651336192f;   // 1/sqrt(97)
    float s[3];
#pragma unroll
    for (int j = 0; j < 3; ++j) {
      const float* kk = &lds_qkv[(e * 3 + j) * QS + E_DIM];
      float d = 0.f;
      for (int c = 0; c < E_DIM; ++c) d += q[c] * kk[c];
      s[j] = d * scale;
    }
    float m = fmaxf(s[0], fmaxf(s[1], s[2]));
    float w0 = __expf(s[0] - m), w1 = __expf(s[1] - m), w2 = __expf(s[2] - m);
    float inv = 1.0f / (w0 + w1 + w2);
    w0 *= inv; w1 *= inv; w2 *= inv;
    const float* v0 = &lds_qkv[(e * 3 + 0) * QS + 2 * E_DIM];
    const float* v1 = &lds_qkv[(e * 3 + 1) * QS + 2 * E_DIM];
    const float* v2 = &lds_qkv[(e * 3 + 2) * QS + 2 * E_DIM];
    unsigned short* outp = Xa + (size_t)(elemBase + e) * XA_STRIDE + i * E_DIM;
    for (int c = 0; c < E_DIM; ++c)
      outp[c] = f2bf(w0 * v0[c] + w1 * v1[c] + w2 * v2[c]);
  }
  // zero K padding cols 291..319
  for (int idx = tid; idx < 16 * (XA_STRIDE - E3); idx += blockDim.x) {
    int e = idx / (XA_STRIDE - E3), c = idx - e * (XA_STRIDE - E3);
    Xa[(size_t)(elemBase + e) * XA_STRIDE + E3 + c] = 0;
  }
}

// ---------------------------------------------------------------------------
// Kernel 3: fused (out-proj ∘ encoder) GEMM (WMMA bf16) + leaky-ReLU +
// anomaly-softmax / prototype-sigmoid heads. 256 threads (8 waves),
// 64 elements per workgroup; A tile TDM-staged into LDS.
// ---------------------------------------------------------------------------
__global__ void k_enc_head(const unsigned short* __restrict__ Xa,
                           const unsigned short* __restrict__ Wc,
                           const float* __restrict__ bcomb,
                           const float* __restrict__ anom_w, const float* __restrict__ anom_b,
                           const float* __restrict__ proto_w, const float* __restrict__ proto_b,
                           float* __restrict__ out) {
  extern __shared__ char smem3[];
  unsigned short* lds_a = (unsigned short*)smem3;             // 64*320*2 = 40960 B @ LDS 0
  float* lds_lat = (float*)(smem3 + 64 * XA_STRIDE * 2);      // 64*160*4 = 40960 B
  int tid = threadIdx.x;
  int elemBase = blockIdx.x * 64;

  if (tid < 32) {
    tdm_load_1d(Xa + (size_t)elemBase * XA_STRIDE, 0u, 64u * XA_STRIDE);
    __builtin_amdgcn_s_wait_tensorcnt(0);
  }
  __syncthreads();

  int lane = tid & 31, wave = tid >> 5;
  int hf = lane >> 4, nlo = lane & 15;

  for (int tile = wave; tile < 40; tile += 8) {
    int mt = tile & 3, nt = tile >> 2;
    int nrow = nt * 16 + nlo;              // < 160 always
    __builtin_prefetch(&Wc[(size_t)nrow * XA_STRIDE], 0, 1);
    v8f acc = {0.f, 0.f, 0.f, 0.f, 0.f, 0.f, 0.f, 0.f};
#pragma unroll
    for (int kt = 0; kt < 10; ++kt) {
      int kb = kt * 32;
      Frag A, Bf;
#pragma unroll
      for (int v = 0; v < 8; ++v) {
        int ka = kb + 2 * (v & 3) + 8 * hf + 16 * (v >> 2);
        A.u[v] = *(const unsigned int*)&lds_a[(mt * 16 + nlo) * XA_STRIDE + ka];
        int kbcol = kb + 2 * (v & 3) + 8 * hf + 16 * (v >> 2);
        Bf.u[v] = *(const unsigned int*)&Wc[(size_t)nrow * XA_STRIDE + kbcol];
      }
      acc = __builtin_amdgcn_wmma_f32_16x16x32_bf16(false, A.v, false, Bf.v,
                                                    (short)0, acc, false, false);
    }
    float bias = bcomb[nrow];
#pragma unroll
    for (int j = 0; j < 8; ++j) {
      int row = mt * 16 + j + 8 * hf;
      float v = acc[j] + bias;
      lds_lat[row * LAT + nrow] = (v > 0.f) ? v : 0.01f * v;   // leaky_relu 0.01
    }
  }
  __syncthreads();

  // heads: 64 elements x 16 latent-heads
  for (int idx = tid; idx < 64 * NH; idx += blockDim.x) {
    int eL = idx >> 4, hh = idx & 15;
    const float* lat = &lds_lat[eL * LAT + hh * NL];
    size_t bgl = (size_t)(elemBase + eL) * NH + hh;
    float s0 = anom_b[0], s1 = anom_b[1];
#pragma unroll
    for (int l = 0; l < NL; ++l) { s0 += lat[l] * anom_w[l]; s1 += lat[l] * anom_w[NL + l]; }
    float m = fmaxf(s0, s1);
    float e0 = __expf(s0 - m), e1 = __expf(s1 - m);
    float inv = 1.0f / (e0 + e1);
    out[bgl * 2 + 0] = e0 * inv;
    out[bgl * 2 + 1] = e1 * inv;
    float* pout = out + (size_t)BATCH * NH * 2 + bgl * 16;
#pragma unroll
    for (int p = 0; p < 16; ++p) {
      float acc2 = proto_b[p];
#pragma unroll
      for (int l = 0; l < NL; ++l) acc2 += lat[l] * proto_w[p * NL + l];
      pout[p] = 1.0f / (1.0f + __expf(-acc2));
    }
  }
}

// ---------------------------------------------------------------------------
extern "C" void kernel_launch(void* const* d_in, const int* in_sizes, int n_in,
                              void* d_out, int out_size, void* d_ws, size_t ws_size,
                              hipStream_t stream) {
  const float* t         = (const float*)d_in[0];
  const float* h0        = (const float*)d_in[1];
  const float* gru_Wih   = (const float*)d_in[2];
  const float* gru_Whh   = (const float*)d_in[3];
  const float* gru_bih   = (const float*)d_in[4];
  const float* gru_bhh   = (const float*)d_in[5];
  const float* gat_fc    = (const float*)d_in[6];
  const float* gat_attn  = (const float*)d_in[7];
  const float* mha_in_w  = (const float*)d_in[8];
  const float* mha_in_b  = (const float*)d_in[9];
  const float* mha_out_w = (const float*)d_in[10];
  const float* mha_out_b = (const float*)d_in[11];
  const float* enc_w     = (const float*)d_in[12];
  const float* enc_b     = (const float*)d_in[13];
  const float* anom_w    = (const float*)d_in[14];
  const float* anom_b    = (const float*)d_in[15];
  const float* proto_w   = (const float*)d_in[16];
  const float* proto_b   = (const float*)d_in[17];
  float* out = (float*)d_out;
  (void)in_sizes; (void)n_in; (void)out_size; (void)ws_size;

  char* ws = (char*)d_ws;
  size_t off = 0;
  auto alloc = [&](size_t bytes) { size_t p = off; off = (off + bytes + 255) & ~(size_t)255; return p; };
  unsigned short* W1    = (unsigned short*)(ws + alloc((size_t)NPAD_QKV * KPAD * 2));
  unsigned short* Wc    = (unsigned short*)(ws + alloc((size_t)LAT * XA_STRIDE * 2));
  float*          bcomb = (float*)         (ws + alloc((size_t)LAT * 4));
  unsigned short* Xc    = (unsigned short*)(ws + alloc((size_t)BATCH * 3 * KPAD * 2));
  unsigned short* Xa    = (unsigned short*)(ws + alloc((size_t)BATCH * XA_STRIDE * 2));

  k_prep<<<128, 256, 0, stream>>>(mha_in_w, mha_out_w, mha_out_b, enc_w, enc_b, W1, Wc, bcomb);
  k_gru_gat<<<BATCH / 256, 256, 0, stream>>>(t, h0, gru_Wih, gru_Whh, gru_bih, gru_bhh,
                                             gat_fc, gat_attn, Xc);
  size_t sh2 = (size_t)48 * KPAD * 2 + (size_t)48 * 292 * 4;   // 68352 B
  k_qkv_attn<<<BATCH / 16, 128, sh2, stream>>>(Xc, W1, mha_in_b, Xa);
  size_t sh3 = (size_t)64 * XA_STRIDE * 2 + (size_t)64 * LAT * 4;  // 81920 B
  k_enc_head<<<BATCH / 64, 256, sh3, stream>>>(Xa, Wc, bcomb, anom_w, anom_b, proto_w, proto_b, out);
}
